// STU_20358144983625
// MI455X (gfx1250) — matrix operand order repaired and checked by reading, not compiled, not run on hardware
//
#include <hip/hip_runtime.h>
#include <hip/hip_bf16.h>
#include <stdint.h>

// Problem constants (from reference)
#define BSZ  4
#define SL   2048
#define DIM  512
#define KEIG 24
#define TT   32      // rows per t-tile / per reduction chunk
#define LDC  520     // cbuf leading dim (u16): 260 dwords % 64 banks = 4 -> conflict-free
#define CHUNK_DW 8192  // dwords per 32x512 bf16 chunk (fragment-major order)

typedef __bf16 bf16_t;
typedef bf16_t        v16bf __attribute__((ext_vector_type(16)));
typedef bf16_t        v2bf  __attribute__((ext_vector_type(2)));
typedef float         v8f   __attribute__((ext_vector_type(8)));
typedef unsigned short ush;
typedef ush           v16us __attribute__((ext_vector_type(16)));
typedef unsigned int  v4ui  __attribute__((ext_vector_type(4)));
typedef int           v8si  __attribute__((ext_vector_type(8)));
typedef int           v4si  __attribute__((ext_vector_type(4)));

// ---------------------------------------------------------------------------
// WMMA helpers (CDNA5 wave32, 16x16x32 bf16 -> f32)
// ---------------------------------------------------------------------------
__device__ __forceinline__ v8f wmma_bf16(v16bf a, v16bf b, v8f c) {
  return __builtin_amdgcn_wmma_f32_16x16x32_bf16(false, a, false, b,
                                                 (short)0, c, false, false);
}

// A-fragment K-slot mapping (ISA 7.12.2, 16-bit A 16x32):
// lanes 0-15: V0..3 -> K=0..7, V4..7 -> K=16..23 ; lanes 16-31: +8
__device__ __forceinline__ int a_k0(int j, int half) {
  return ((j < 4) ? (2 * j) : (2 * j + 8)) + half * 8;
}

// A fragment from reversed Hankel filter table: element (M,K) = h[base + K]
// (base folds in per-lane row M; odd bases -> u16 gathers, amortized 1:8 WMMAs)
__device__ __forceinline__ v16bf load_a_hankel(const ush* h, int base, int half) {
  v16us t;
#pragma unroll
  for (int j = 0; j < 8; ++j) {
    const int k0 = a_k0(j, half);
    t[2 * j]     = h[base + k0];
    t[2 * j + 1] = h[base + k0 + 1];
  }
  return __builtin_bit_cast(v16bf, t);
}

// A fragment from a row-major LDS tile: K-slots j0..3 and j4..7 are each
// 8 contiguous u16 (at koff+8*half and koff+16+8*half) -> two ds_load_b128.
__device__ __forceinline__ v16bf load_a_row16(const bf16_t* buf, int row, int ld,
                                              int koff, int half) {
  const v4ui* p = (const v4ui*)(buf + row * ld + koff + 8 * half);
  struct S { v4ui a, b; } s;
  s.a = p[0];      // K slots j=0..3
  s.b = p[2];      // +16 u16: K slots j=4..7
  return __builtin_bit_cast(v16bf, s);
}

// B fragment from a fragment-major staged chunk: lane's fragment is two
// contiguous 16B pieces -> two ds_load_b128, bank-conflict-free at floor.
// Chunk layout (dwords): nt*256 + piece*128 + lane*4 + c
__device__ __forceinline__ v16bf load_b_frag(const v4ui* buf, int nt, int lane) {
  struct S { v4ui a, b; } s;
  s.a = buf[nt * 64 + lane];        // piece 0: VGPR j=0..3
  s.b = buf[nt * 64 + 32 + lane];   // piece 1: VGPR j=4..7
  return __builtin_bit_cast(v16bf, s);
}

// ---------------------------------------------------------------------------
// Tensor Data Mover: contiguous chunk global -> LDS (1D tile).
// D# packed per CDNA5 ISA ch.8; clang-23 6-arg builtin.
// ---------------------------------------------------------------------------
__device__ __forceinline__ void tdm_load_chunk(unsigned lds_off, const void* gptr,
                                               unsigned elems /* u16, <=65535 */) {
  const unsigned long long ga = (unsigned long long)(uintptr_t)gptr;
  v4ui g0;
  g0[0] = 1u;                                        // count=1 (valid user D#)
  g0[1] = lds_off;                                   // lds_addr [63:32]
  g0[2] = (unsigned)(ga & 0xFFFFFFFFu);              // global_addr lo
  g0[3] = (unsigned)((ga >> 32) & 0x01FFFFFFu) | (2u << 30);  // addr hi | type=2
  v8si g1;
  g1[0] = (int)(1u << 16);                           // data_size=1 (2B)
  g1[1] = (int)((elems & 0xFFFFu) << 16);            // tensor_dim0 lo16 @bit48
  g1[2] = (int)(((elems >> 16) & 0xFFFFu) | (1u << 16)); // dim0 hi | tensor_dim1=1
  g1[3] = (int)((elems & 0xFFFFu) << 16);            // dim1 hi=0 | tile_dim0=elems
  g1[4] = 1;                                         // tile_dim1=1 (tile_dim2=0)
  g1[5] = (int)elems;                                // tensor_dim0_stride lo32
  g1[6] = 0;
  g1[7] = 0;
  v4si z4 = {0, 0, 0, 0};
  v8si z8 = {0, 0, 0, 0, 0, 0, 0, 0};
  __builtin_amdgcn_tensor_load_to_lds(g0, g1, z4, z4, z8, 0);
}

// ---------------------------------------------------------------------------
// Prep kernels: bf16 convert + permute each 32x512 chunk into fragment-major
// order. Thread o produces one dword = rows {2p,2p+1} of one column, where
// o = base + nt*256 + P*128 + (h*16+ln)*4 + c4,  p = 4h + P*8 + c4.
// ---------------------------------------------------------------------------
__device__ __forceinline__ void frag_decode(int r2, int& p, int& col) {
  int nt = r2 / 256, r3 = r2 % 256;
  int P  = r3 / 128, r4 = r3 % 128;
  int pi = r4 / 4,   c4 = r4 % 4;
  int h = pi / 16, ln = pi % 16;
  p   = 4 * h + P * 8 + c4;      // row-pair within chunk
  col = nt * 16 + ln;
}

__global__ void stu_prep_u(const float* __restrict__ in,
                           unsigned* __restrict__ Ubf) {
  int o = blockIdx.x * 256 + threadIdx.x;            // dword index
  if (o >= BSZ * SL * DIM / 2) return;
  const int per_b = SL * DIM / 2;
  int b = o / per_b, r = o % per_b;
  int chunk = r / CHUNK_DW, r2 = r % CHUNK_DW;
  int p, col;
  frag_decode(r2, p, col);
  int row = chunk * TT + 2 * p;
  float lo = in[((size_t)b * SL + row) * DIM + col];
  float hi = in[((size_t)b * SL + row + 1) * DIM + col];
  v2bf pk = {(bf16_t)lo, (bf16_t)hi};                // hw cvt f32->bf16 (RNE)
  Ubf[o] = __builtin_bit_cast(unsigned, pk);
}

__global__ void stu_prep_w(const float* __restrict__ m_phi,
                           const float* __restrict__ eig_vals,
                           unsigned* __restrict__ Wbf) {
  int o = blockIdx.x * 256 + threadIdx.x;
  if (o >= KEIG * DIM * DIM / 2) return;
  const int per_k = DIM * DIM / 2;
  int k = o / per_k, r = o % per_k;
  int chunk = r / CHUNK_DW, r2 = r % CHUNK_DW;
  int p, col;
  frag_decode(r2, p, col);
  int row = chunk * TT + 2 * p;                      // d index in m_phi_k
  float s = sqrtf(sqrtf(eig_vals[k]));               // eig^0.25 folded into W
  float lo = m_phi[((size_t)k * DIM + row) * DIM + col] * s;
  float hi = m_phi[((size_t)k * DIM + row + 1) * DIM + col] * s;
  v2bf pk = {(bf16_t)lo, (bf16_t)hi};
  Wbf[o] = __builtin_bit_cast(unsigned, pk);
}

// Hbf[k][i] = v_k[2047-i] for i<2048, else 0  => Toeplitz T[t,s]=Hbf[2047-t+s]
__global__ void stu_prep_h(const float* __restrict__ eig_vecs, ush* __restrict__ Hbf) {
  int i = blockIdx.x * 256 + threadIdx.x;
  if (i >= KEIG * 2 * SL) return;
  int k = i / (2 * SL);
  int j = i % (2 * SL);
  float v = (j < SL) ? eig_vecs[(SL - 1 - j) * KEIG + k] : 0.0f;
  bf16_t bv = (bf16_t)v;
  Hbf[i] = __builtin_bit_cast(ush, bv);
}

// ---------------------------------------------------------------------------
// Fused main kernel: per (32-row t-tile, batch) workgroup; 8 waves 2(t)x4(e),
// each wave owns 16x128 fragments. For each k:
//   phase A: conv = T_k @ U   (Toeplitz A from LDS filter, TDM-staged U)
//   phase B: out += conv @ W_k (TDM-staged W, f32 out resident in VGPRs)
// Both phases double-buffer the shared 2x32KB staging buffer via TENSORcnt.
// ---------------------------------------------------------------------------
__global__ void __launch_bounds__(256)
stu_main(const ush* __restrict__ Ubf, const ush* __restrict__ Wbf,
         const ush* __restrict__ Hbf, float* __restrict__ out) {
  __shared__ __align__(16) ush    sbuf[2][TT * DIM];  // 2x32 KB staging chunks
  __shared__ __align__(16) bf16_t cbuf[TT * LDC];     // 33 KB conv tile (padded)
  __shared__ __align__(16) ush    hbuf[2 * SL];       //  8 KB reversed filter k

  const int tid  = threadIdx.x;
  const int lane = tid & 31, wave = tid >> 5;
  const int wt = wave >> 2;          // 0..1 : 16-row slice
  const int we = wave & 3;           // 0..3 : 128-col slice
  const int half = lane >> 4, ln = lane & 15;
  const int t0 = blockIdx.x * TT;
  const int b  = blockIdx.y;
  const int niterA = t0 / TT + 1;    // causal s-blocks
  const int niterB = DIM / TT;       // 16 reduction chunks

  v8f oacc[8];
#pragma unroll
  for (int nt = 0; nt < 8; ++nt) oacc[nt] = (v8f){0.f,0.f,0.f,0.f,0.f,0.f,0.f,0.f};

  for (int k = 0; k < KEIG; ++k) {
    // stage reversed filter for this k (8 KB, cooperative vector copy)
    __syncthreads();                                 // prior k fully done
    {
      const v4ui* src = (const v4ui*)(Hbf + (size_t)k * (2 * SL));
      v4ui* dst = (v4ui*)hbuf;
      dst[tid]       = src[tid];
      dst[tid + 256] = src[tid + 256];
    }

    v8f cacc[8];
#pragma unroll
    for (int nt = 0; nt < 8; ++nt) cacc[nt] = (v8f){0.f,0.f,0.f,0.f,0.f,0.f,0.f,0.f};

    // ---- Phase A: conv[t,d] = sum_{s<=t} v_k[t-s] * u[s,d]
    if (wave == 0)                                   // prologue: stage s-block 0
      tdm_load_chunk((unsigned)(uintptr_t)&sbuf[0][0],
                     Ubf + (size_t)b * SL * DIM, TT * DIM);
    for (int i = 0; i < niterA; ++i) {
      if (wave == 0) {
        if (i + 1 < niterA) {                        // prefetch next s-block
          tdm_load_chunk((unsigned)(uintptr_t)&sbuf[(i + 1) & 1][0],
                         Ubf + (size_t)b * SL * DIM + (size_t)(i + 1) * TT * DIM,
                         TT * DIM);
          __builtin_amdgcn_s_wait_tensorcnt(1);      // tile i done, i+1 in flight
        } else {
          __builtin_amdgcn_s_wait_tensorcnt(0);      // last tile done
        }
      }
      __syncthreads();                               // tile i visible (+hbuf on i==0)
      const v4ui* ub = (const v4ui*)&sbuf[i & 1][0];
      const int s0 = i * TT;
      const int abase = 2047 - (t0 + wt * 16) - ln + s0;  // s>t lands in zero pad
      v16bf af = load_a_hankel(hbuf, abase, half);
#pragma unroll
      for (int nt = 0; nt < 8; ++nt) {
        v16bf bf = load_b_frag(ub, we * 8 + nt, lane);
        cacc[nt] = wmma_bf16(af, bf, cacc[nt]);
      }
      __syncthreads();                               // readers done before rewrite
    }

    // publish conv tile as bf16 (scale lives in W)
#pragma unroll
    for (int nt = 0; nt < 8; ++nt)
#pragma unroll
      for (int r = 0; r < 8; ++r)
        cbuf[(wt * 16 + half * 8 + r) * LDC + we * 128 + nt * 16 + ln] =
            (bf16_t)cacc[nt][r];

    // ---- Phase B: out += conv @ W_k  (reduction over d in 32-chunks)
    if (wave == 0)                                   // prologue: stage d-chunk 0
      tdm_load_chunk((unsigned)(uintptr_t)&sbuf[0][0],
                     Wbf + (size_t)k * DIM * DIM, TT * DIM);
    for (int i = 0; i < niterB; ++i) {
      if (wave == 0) {
        if (i + 1 < niterB) {
          tdm_load_chunk((unsigned)(uintptr_t)&sbuf[(i + 1) & 1][0],
                         Wbf + (size_t)k * DIM * DIM + (size_t)(i + 1) * TT * DIM,
                         TT * DIM);
          __builtin_amdgcn_s_wait_tensorcnt(1);
        } else {
          __builtin_amdgcn_s_wait_tensorcnt(0);
        }
      }
      __syncthreads();                               // tile visible (+cbuf on i==0)
      const v4ui* wb = (const v4ui*)&sbuf[i & 1][0];
      const int d0 = i * TT;
      v16bf af = load_a_row16(cbuf, wt * 16 + ln, LDC, d0, half);
#pragma unroll
      for (int nt = 0; nt < 8; ++nt) {
        v16bf bf = load_b_frag(wb, we * 8 + nt, lane);
        oacc[nt] = wmma_bf16(af, bf, oacc[nt]);
      }
      __syncthreads();
    }
  }

  // write f32 output (C/D layout: M = r + 8*half, N = ln)
#pragma unroll
  for (int nt = 0; nt < 8; ++nt)
#pragma unroll
    for (int r = 0; r < 8; ++r) {
      int row = t0 + wt * 16 + half * 8 + r;
      int col = we * 128 + nt * 16 + ln;
      out[((size_t)b * SL + row) * DIM + col] = oacc[nt][r];
    }
}

// ---------------------------------------------------------------------------
extern "C" void kernel_launch(void* const* d_in, const int* in_sizes, int n_in,
                              void* d_out, int out_size, void* d_ws, size_t ws_size,
                              hipStream_t stream) {
  const float* inputs  = (const float*)d_in[0];   // [4,2048,512] f32
  const float* eigvals = (const float*)d_in[1];   // [24] f32
  const float* eigvecs = (const float*)d_in[2];   // [2048,24] f32
  const float* mphi    = (const float*)d_in[3];   // [12288,512] f32
  float* out = (float*)d_out;                     // [4,2048,512] f32

  // Workspace (bf16, fragment-major chunks): W 12.6MB | U 8.4MB | H 0.19MB
  char* ws = (char*)d_ws;
  ush* Wbf = (ush*)ws;
  ush* Ubf = (ush*)(ws + (size_t)KEIG * DIM * DIM * 2);
  ush* Hbf = (ush*)(ws + (size_t)KEIG * DIM * DIM * 2 + (size_t)BSZ * SL * DIM * 2);

  stu_prep_w<<<(KEIG * DIM * DIM / 2 + 255) / 256, 256, 0, stream>>>(
      mphi, eigvals, (unsigned*)Wbf);
  stu_prep_u<<<(BSZ * SL * DIM / 2 + 255) / 256, 256, 0, stream>>>(
      inputs, (unsigned*)Ubf);
  stu_prep_h<<<(KEIG * 2 * SL + 255) / 256, 256, 0, stream>>>(eigvecs, Hbf);

  dim3 grid(SL / TT, BSZ);   // 64 t-tiles x 4 batches = 256 workgroups
  stu_main<<<grid, 256, 0, stream>>>(Ubf, Wbf, Hbf, out);
}
